// SpatialCorrelationSampler_42949672961273
// MI455X (gfx1250) — matrix operand order repaired and checked by reading, not compiled, not run on hardware
//
#include <hip/hip_runtime.h>

// MI455X / gfx1250 spatial correlation sampler (P=21, dilation=1).
// Banded-GEMM: per (b, y, dy), G[x, x'] = sum_c in1[c,x] * in2[c,x'] via
// v_wmma_f32_16x16x32_bf16 with 3xBF16 hi/lo split (~fp32 accuracy); only the
// |x'-x|<=10 band is kept (3 N-tiles per M-tile). A fragments live in
// registers across all 21 dy. in2 K-chunks are double-buffered in LDS with
// branch-free packed staging (v_cvt_pk_bf16_f32 pairs + v_fma_mix_f32_bf16
// residuals via inline asm); output staged through LDS and written with
// non-temporal stores so the streaming 116MB output does not evict the
// L2-resident 128MB input set.

typedef __attribute__((ext_vector_type(16))) __bf16        v16bf;
typedef __attribute__((ext_vector_type(8)))  float         v8f;
typedef __attribute__((ext_vector_type(4)))  unsigned int  v4u;
typedef __attribute__((ext_vector_type(2)))  unsigned int  v2u;
typedef __attribute__((ext_vector_type(2)))  float         v2f;
typedef __attribute__((ext_vector_type(2)))  __bf16        v2bf;

namespace {
constexpr int kB = 4, kC = 256, kH = 128, kW = 128, kP = 21;
constexpr int kHW = kH * kW;
constexpr int kChunk = 32;        // channels per K-step group (8 chunks)
constexpr int kS  = 40;           // LDS row stride (bf16): 80B, 16B-aligned, padded
constexpr int kX2 = kW + 32;      // in2 x'' extent: 16 zero-pad columns each side

// LDS layout inside one flat block (all offsets 16B aligned):
//   [0      , 12800) s2h buf0      [12800 , 25600) s2l buf0
//   [25600  , 38400) s2h buf1      [38400 , 51200) s2l buf1
//   [51200  , 61952) s_out (21*128 f32)
// phase-0 alias (dead after A->registers): s1h @25600 (10240B), s1l @35840 (10240B)
constexpr int kSmemBytes = 61952;
}

union Frag { v4u q[2]; v16bf v; };

// residual = r - (float)bf16_half(pk), computed in ONE op via
// v_fma_mix_f32_bf16 (op_sel selects the 16-bit half of src0):
//   {op_sel_hi,op_sel} = {1,0} -> src[15:0],  {1,1} -> src[31:16]
__device__ __forceinline__ float fmix_res_lo(unsigned pk, float r) {
  float d;
  asm("v_fma_mix_f32_bf16 %0, %1, -1.0, %2 op_sel:[0,0,0] op_sel_hi:[1,0,0]"
      : "=v"(d) : "v"(pk), "v"(r));
  return d;
}
__device__ __forceinline__ float fmix_res_hi(unsigned pk, float r) {
  float d;
  asm("v_fma_mix_f32_bf16 %0, %1, -1.0, %2 op_sel:[1,0,0] op_sel_hi:[1,0,0]"
      : "=v"(d) : "v"(pk), "v"(r));
  return d;
}

// Split 4 fp32 -> packed bf16 hi + lo pairs. __builtin_convertvector on
// float2 -> bf16x2 maps onto two-source v_cvt_pk_bf16_f32; residuals use
// v_fma_mix_f32_bf16 directly. ~2 VALU per element total.
__device__ __forceinline__ void split4(const float r[4], v2u& hv, v2u& lv) {
  union PB { v2bf h; unsigned u; };
  v2f a01; a01.x = r[0]; a01.y = r[1];
  v2f a23; a23.x = r[2]; a23.y = r[3];
  PB h01, h23;
  h01.h = __builtin_convertvector(a01, v2bf);
  h23.h = __builtin_convertvector(a23, v2bf);
  v2f e01, e23;
  e01.x = fmix_res_lo(h01.u, r[0]);
  e01.y = fmix_res_hi(h01.u, r[1]);
  e23.x = fmix_res_lo(h23.u, r[2]);
  e23.y = fmix_res_hi(h23.u, r[3]);
  PB l01, l23;
  l01.h = __builtin_convertvector(e01, v2bf);
  l23.h = __builtin_convertvector(e23, v2bf);
  hv.x = h01.u; hv.y = h23.u;
  lv.x = l01.u; lv.y = l23.u;
}

// Convert 16 staged floats (4 c-groups of 4) and store as b64 packed bf16.
__device__ __forceinline__ void stage_store(__bf16* sh, __bf16* sl,
                                            const float r[16], int row, int hc) {
#pragma unroll
  for (int j = 0; j < 4; ++j) {
    v2u hv, lv;
    split4(&r[4 * j], hv, lv);
    const int c0 = (hc + 2 * j) * 4;
    *(v2u*)&sh[row * kS + c0] = hv;
    *(v2u*)&sl[row * kS + c0] = lv;
  }
}

// 16 coalesced global loads with immediate channel offsets (c*HW*4 <= 1.8MB).
__device__ __forceinline__ void load16(float r[16], const float* __restrict__ p) {
#pragma unroll
  for (int j = 0; j < 4; ++j)
#pragma unroll
    for (int i = 0; i < 4; ++i)
      r[j * 4 + i] = p[(size_t)(8 * j + i) * kHW];
}

__global__ __launch_bounds__(256)
void corr21_wmma_bf16x3(const float* __restrict__ in1,
                        const float* __restrict__ in2,
                        float* __restrict__ out)
{
  __shared__ __align__(16) char smem[kSmemBytes];
  __bf16* const s2h0  = (__bf16*)(smem);
  __bf16* const s2l0  = (__bf16*)(smem + 12800);
  __bf16* const s2h1  = (__bf16*)(smem + 25600);
  __bf16* const s2l1  = (__bf16*)(smem + 38400);
  float*  const s_out = (float*) (smem + 51200);
  __bf16* const s1h   = (__bf16*)(smem + 25600);  // phase-0 alias over buf1
  __bf16* const s1l   = (__bf16*)(smem + 35840);

  const int tid  = threadIdx.x;
  const int lane = tid & 31;
  const int wv   = tid >> 5;      // wave id == M-tile index (x in [16*wv, 16*wv+16))
  const int ln   = lane & 15;     // A: row m / B: column n
  const int half = lane >> 4;     // lane half -> K sub-range per ISA layouts
  const int y    = blockIdx.x;
  const int b    = blockIdx.y;
  const int x    = tid & 127;     // staging column (interior coordinate)
  const int hc   = tid >> 7;      // staging channel-half (0/1)

  // ---------------- Phase 0: in1 row -> per-wave A fragments (hi/lo) in
  // registers; reused from registers across all 21 dy.
  v16bf Ah[8], Al[8];
  {
    const float* p1 = in1 + (size_t)b * kC * kHW + (size_t)y * kW + x
                          + (size_t)hc * 4 * kHW;
    const int arow = (wv * 16 + ln) * kS;
#pragma unroll
    for (int ch = 0; ch < 8; ++ch) {
      __syncthreads();
      float r[16];
      load16(r, p1);
      stage_store(s1h, s1l, r, x, hc);
      p1 += (size_t)kChunk * kHW;
      __syncthreads();
      // 16-bit A 16x32: lanes 0-15 hold K = 0..7 & 16..23,
      // lanes 16-31 hold K = 8..15 & 24..31 -> two 16B groups per lane.
      Frag fh, fl;
      fh.q[0] = *(const v4u*)&s1h[arow + 8 * half];
      fh.q[1] = *(const v4u*)&s1h[arow + 16 + 8 * half];
      fl.q[0] = *(const v4u*)&s1l[arow + 8 * half];
      fl.q[1] = *(const v4u*)&s1l[arow + 16 + 8 * half];
      Ah[ch] = fh.v;
      Al[ch] = fl.v;
    }
  }

  // ---------------- One-time zero-fill of the +-16 pad columns of BOTH in2
  // buffers (they alias s1, so this must come after phase 0; __syncthreads
  // drains DScnt before the barrier, ordering the aliased accesses).
  __syncthreads();
  {
    const int row = tid >> 3;                 // 0..31 pad rows
    const int c0  = (tid & 7) * 4;            // 8 groups of 4 channels
    const int xq  = (row < 16) ? row : row + 128;  // [0,16) U [144,160)
    v2u z; z.x = 0u; z.y = 0u;
    *(v2u*)&s2h0[xq * kS + c0] = z;
    *(v2u*)&s2l0[xq * kS + c0] = z;
    *(v2u*)&s2h1[xq * kS + c0] = z;
    *(v2u*)&s2l1[xq * kS + c0] = z;
  }

  // ---------------- Main loop over vertical displacement dy = ph - 10.
  for (int ph = 0; ph < kP; ++ph) {
    const int y2 = y + ph - (kP / 2);
    const size_t obase = (size_t)(b * kP + ph) * kP * kHW + (size_t)y * kW;

    if (y2 < 0 || y2 >= kH) {     // uniform branch: whole slab is zeros
      for (int idx = tid; idx < kP * kW; idx += 256) {
        int pw = idx >> 7;
        int xx = idx & (kW - 1);
        __builtin_nontemporal_store(0.0f, &out[obase + (size_t)pw * kHW + xx]);
      }
      continue;
    }

    const float* p2 = in2 + (size_t)b * kC * kHW + (size_t)y2 * kW + x
                          + (size_t)hc * 4 * kHW;
    float r[16];

    // Prologue: stage chunk 0 into buf0 (pads already zero, interior only).
    load16(r, p2);
    stage_store(s2h0, s2l0, r, x + 16, hc);
    p2 += (size_t)kChunk * kHW;

    v8f acc[3] = {};              // 3 band N-tiles, f32 accumulators

#pragma unroll
    for (int ch = 0; ch < 8; ++ch) {
      __syncthreads();            // buf[ch&1] stores visible; old buffer free
      if (ch < 7) {               // issue next chunk's loads early (overlap)
        load16(r, p2);
        p2 += (size_t)kChunk * kHW;
      }
      const __bf16* bh = (ch & 1) ? s2h1 : s2h0;
      const __bf16* bl = (ch & 1) ? s2l1 : s2l0;
#pragma unroll
      for (int t = 0; t < 3; ++t) {
        // 16-bit B 32x16: lanes 0-15 (col n) hold K=0..7 & 16..23, lanes 16-31
        // hold K=8..15 & 24..31 -> two 16B groups per lane (matches A).
        const int br = (16 * wv + 16 * t + ln) * kS + 8 * half;
        Frag fbh, fbl;
        fbh.q[0] = *(const v4u*)&bh[br];
        fbh.q[1] = *(const v4u*)&bh[br + 16];
        fbl.q[0] = *(const v4u*)&bl[br];
        fbl.q[1] = *(const v4u*)&bl[br + 16];
        // 3xBF16 fp32 emulation: hi*hi + hi*lo + lo*hi
        acc[t] = __builtin_amdgcn_wmma_f32_16x16x32_bf16(
            false, Ah[ch], false, fbh.v, (short)0, acc[t], false, false);
        acc[t] = __builtin_amdgcn_wmma_f32_16x16x32_bf16(
            false, Ah[ch], false, fbl.v, (short)0, acc[t], false, false);
        acc[t] = __builtin_amdgcn_wmma_f32_16x16x32_bf16(
            false, Al[ch], false, fbh.v, (short)0, acc[t], false, false);
      }
      if (ch < 7) {               // convert + store into the other buffer
        __bf16* nh = (ch & 1) ? s2h0 : s2h1;
        __bf16* nl = (ch & 1) ? s2l0 : s2l1;
        stage_store(nh, nl, r, x + 16, hc);
      }
    }

    // Extract the valid band into LDS staging [pw][x].
    // C/D layout: VGPR v at lane l holds (m = v + 8*(l>>4), n = l&15);
    // x'' = 16*wv + 16*t + n  ->  pw = 16*t + n - m - 6.
#pragma unroll
    for (int t = 0; t < 3; ++t) {
#pragma unroll
      for (int v = 0; v < 8; ++v) {
        int m  = v + 8 * half;
        int pw = 16 * t + ln - m - 6;
        if (pw >= 0 && pw < kP) s_out[pw * kW + 16 * wv + m] = acc[t][v];
      }
    }
    __syncthreads();

    // Coalesced, non-temporal global stores: each pw row is 512 contiguous
    // bytes; NT keeps the streaming output from evicting L2-resident inputs.
    for (int idx = tid; idx < kP * kW; idx += 256) {
      int pw = idx >> 7;
      int xx = idx & (kW - 1);
      __builtin_nontemporal_store(s_out[idx], &out[obase + (size_t)pw * kHW + xx]);
    }
    // next iteration's barriers protect s_out / staging buffer reuse
  }
}

extern "C" void kernel_launch(void* const* d_in, const int* in_sizes, int n_in,
                              void* d_out, int out_size, void* d_ws, size_t ws_size,
                              hipStream_t stream) {
  (void)in_sizes; (void)n_in; (void)out_size; (void)d_ws; (void)ws_size;
  const float* in1 = (const float*)d_in[0];
  const float* in2 = (const float*)d_in[1];
  float* out = (float*)d_out;
  dim3 grid(kH, kB);   // one workgroup per (y, b); 8 waves = 8 M-tiles of x
  corr21_wmma_bf16x3<<<grid, 256, 0, stream>>>(in1, in2, out);
}
